// EvenOddFunction_71871982731803
// MI455X (gfx1250) — matrix operand order repaired and checked by reading, not compiled, not run on hardware
//
#include <hip/hip_runtime.h>

// ---------------------------------------------------------------------------
// EvenOdd network step: two 16384x2048x2048 GEMMs (f16 WMMA, f32 accum) with
// fused bias/Ux + 10-step pointwise recurrence epilogues.
// CDNA5 data path: GLOBAL_LOAD_ASYNC_TO_LDS_B128 (ASYNCcnt) double-buffered
// LDS tiles feeding v_wmma_f32_16x16x32_f16.
// ---------------------------------------------------------------------------

typedef _Float16 f16x16 __attribute__((ext_vector_type(16)));
typedef _Float16 f16x8  __attribute__((ext_vector_type(8)));
typedef _Float16 f16x4  __attribute__((ext_vector_type(4)));
typedef float    f32x8  __attribute__((ext_vector_type(8)));
typedef float    f32x4  __attribute__((ext_vector_type(4)));

#define B_ROWS 16384
#define SE     2048
#define SO     2048
#define H1     1024
#define SDIM   4096      // SE + SO
#define KDIM   2048
#define NDIM   2048
#define TM     128
#define TN     128
#define TK     64
#define LDP    72        // padded LDS pitch in halves (TK + 8) -> conflict-free

union FragAB {
    f16x16 v;
    f16x8  h[2];
};

__device__ __forceinline__ float rho_f(float x) {
    // sigmoid(4*(x-0.5)) = 1/(1+exp(2-4x))
    return 1.0f / (1.0f + __expf(2.0f - 4.0f * x));
}

// One 16-byte async global->LDS copy (CDNA5, tracked by ASYNCcnt).
// GVS addressing: uniform 64-bit SGPR base + per-lane 32-bit byte offset.
__device__ __forceinline__ void async_copy_b128(unsigned ldsAddr,
                                                unsigned gvOff,
                                                const void* sBase) {
    asm volatile("global_load_async_to_lds_b128 %0, %1, %2"
                 :
                 : "v"(ldsAddr), "v"(gvOff), "s"(sBase)
                 : "memory");
}

__device__ __forceinline__ void wait_async0() {
    asm volatile("s_wait_asynccnt 0" ::: "memory");
}

// A1[b, k] = f16(rho(s_even[b, k])), vectorized x4
__global__ __launch_bounds__(256)
void prep_rho_kernel(const float* __restrict__ s, _Float16* __restrict__ A1) {
    int gid = blockIdx.x * blockDim.x + threadIdx.x;
    int i   = gid * 4;                       // flat index into (B_ROWS, SE)
    int row = i >> 11;                       // /2048
    int col = i & 2047;
    f32x4 sv = *(const f32x4*)(s + (size_t)row * SDIM + col);
    f16x4 o;
    o.x = (_Float16)rho_f(sv.x);
    o.y = (_Float16)rho_f(sv.y);
    o.z = (_Float16)rho_f(sv.z);
    o.w = (_Float16)rho_f(sv.w);
    *(f16x4*)(A1 + i) = o;
}

// W16 = f16(W) row-major (SE x SO); Wt16 = f16(W^T) row-major (SO x SE)
__global__ __launch_bounds__(256)
void prep_w_kernel(const float* __restrict__ W,
                   _Float16* __restrict__ W16,
                   _Float16* __restrict__ Wt16) {
    int gid = blockIdx.x * blockDim.x + threadIdx.x;
    int i   = gid * 4;
    int r   = i >> 11;
    int c   = i & 2047;
    f32x4 w = *(const f32x4*)(W + i);
    f16x4 o;
    o.x = (_Float16)w.x;
    o.y = (_Float16)w.y;
    o.z = (_Float16)w.z;
    o.w = (_Float16)w.w;
    *(f16x4*)(W16 + i) = o;
    Wt16[(size_t)(c + 0) * SE + r] = o.x;
    Wt16[(size_t)(c + 1) * SE + r] = o.y;
    Wt16[(size_t)(c + 2) * SE + r] = o.z;
    Wt16[(size_t)(c + 3) * SE + r] = o.w;
}

// C[m,n] = sum_k A[m,k] * Brm[n,k] + bias[n] (+ Ux[m,n] if n < H1 and Ux),
// then 10 recurrence steps on state read from s[:, sColOff + n], result
// written to out[:, outColOff + n]; optionally Anext[m,n] = f16(rho(state)).
__global__ __launch_bounds__(256)
void gemm_iter_kernel(const _Float16* __restrict__ A,     // M x K (f16)
                      const _Float16* __restrict__ Brm,   // N x K (f16)
                      const float* __restrict__ bias,     // N
                      const float* __restrict__ Ux,       // M x H1 or nullptr
                      const float* __restrict__ s,        // M x SDIM
                      float* __restrict__ out,            // M x SDIM
                      _Float16* __restrict__ Anext,       // M x N or nullptr
                      int sColOff, int outColOff) {
    __shared__ _Float16 lA[2][TM * LDP];   // 2 x 18 KB
    __shared__ _Float16 lB[2][TN * LDP];   // 2 x 18 KB  (72 KB total, <320KB/WGP)

    const int tid  = threadIdx.x;
    const int lane = tid & 31;
    const int wave = tid >> 5;      // 0..7
    const int wm   = wave >> 1;     // 0..3 : 32-row strip
    const int wn   = wave & 1;      // 0..1 : 64-col strip

    const int bm    = blockIdx.x & 127;   // M / TM = 128
    const int bn    = blockIdx.x >> 7;    // N / TN = 16
    const int mBase = bm * TM;
    const int nBase = bn * TN;

    // global->LDS staging geometry: 8 threads x 16B per row, 32 rows per pass
    const int ldr = tid >> 3;             // 0..31
    const int ldk = (tid & 7) << 3;       // half offset 0..56

    // Per-lane global byte offsets (advance by TK*2 = 128B each K-tile)
    unsigned voffA = ((unsigned)(mBase + ldr) * KDIM + ldk) * 2u;
    unsigned voffB = ((unsigned)(nBase + ldr) * KDIM + ldk) * 2u;

    // Per-lane LDS byte addresses for each buffer (low 32 bits of flat addr)
    unsigned ldsA[2], ldsB[2];
#pragma unroll
    for (int b = 0; b < 2; ++b) {
        ldsA[b] = (unsigned)(unsigned long long)&lA[b][ldr * LDP + ldk];
        ldsB[b] = (unsigned)(unsigned long long)&lB[b][ldr * LDP + ldk];
    }

    f32x8 acc[2][4];
#pragma unroll
    for (int i = 0; i < 2; ++i)
#pragma unroll
        for (int j = 0; j < 4; ++j) acc[i][j] = (f32x8)0.0f;

    // WMMA fragment addressing (16-bit A 16x32 layout; B 32x16 layout)
    const int arow = wm * 32 + (lane & 15);
    const int akb  = (lane >> 4) << 3;    // +8 halves for upper half-wave
    const int bcol = wn * 64 + (lane & 15);
    const int bkb  = (lane >> 4) << 3;    // +8 halves for upper half-wave

    // --- prologue: async-load K-tile 0 into buffer 0 --------------------
#pragma unroll
    for (int p = 0; p < 4; ++p) {
        async_copy_b128(ldsA[0] + (unsigned)p * 32u * LDP * 2u,
                        voffA + (unsigned)p * 32u * KDIM * 2u, A);
        async_copy_b128(ldsB[0] + (unsigned)p * 32u * LDP * 2u,
                        voffB + (unsigned)p * 32u * KDIM * 2u, Brm);
    }
    voffA += TK * 2u;
    voffB += TK * 2u;
    wait_async0();
    __syncthreads();

    const int KT = KDIM / TK;             // 32
    for (int kt = 0; kt < KT; ++kt) {
        const int buf = kt & 1;

        // async-prefetch next K-tile into the other buffer (hides behind WMMA)
        if (kt + 1 < KT) {
            const int nb = buf ^ 1;
#pragma unroll
            for (int p = 0; p < 4; ++p) {
                async_copy_b128(ldsA[nb] + (unsigned)p * 32u * LDP * 2u,
                                voffA + (unsigned)p * 32u * KDIM * 2u, A);
                async_copy_b128(ldsB[nb] + (unsigned)p * 32u * LDP * 2u,
                                voffB + (unsigned)p * 32u * KDIM * 2u, Brm);
            }
            voffA += TK * 2u;
            voffB += TK * 2u;
        }

        // two k-steps of 32 over the current LDS tile
#pragma unroll
        for (int ks = 0; ks < 2; ++ks) {
            const int kb = ks * 32;
            FragAB a[2], b[4];
#pragma unroll
            for (int i = 0; i < 2; ++i) {
                const _Float16* base = &lA[buf][(arow + i * 16) * LDP + kb + akb];
                a[i].h[0] = *(const f16x8*)(base);
                a[i].h[1] = *(const f16x8*)(base + 16);
            }
#pragma unroll
            for (int j = 0; j < 4; ++j) {
                const _Float16* base = &lB[buf][(bcol + j * 16) * LDP + kb + bkb];
                b[j].h[0] = *(const f16x8*)(base);
                b[j].h[1] = *(const f16x8*)(base + 16);
            }
#pragma unroll
            for (int i = 0; i < 2; ++i)
#pragma unroll
                for (int j = 0; j < 4; ++j)
                    acc[i][j] = __builtin_amdgcn_wmma_f32_16x16x32_f16(
                        false, a[i].v, false, b[j].v,
                        (short)0, acc[i][j], false, false);
        }

        // own async issues complete + all waves past their waits
        wait_async0();
        __syncthreads();
    }

    // Fused epilogue: bias (+Ux), 10 recurrence steps, store state (+rho->f16)
#pragma unroll
    for (int j = 0; j < 4; ++j) {
        const int col = nBase + wn * 64 + j * 16 + (lane & 15);
        const float bc = bias[col];
#pragma unroll
        for (int i = 0; i < 2; ++i) {
            const int row0 = mBase + wm * 32 + i * 16 + ((lane >> 4) << 3);
#pragma unroll
            for (int r = 0; r < 8; ++r) {
                const int row = row0 + r;
                float c = acc[i][j][r] + bc;
                if (Ux != nullptr && col < H1)
                    c += Ux[(size_t)row * H1 + col];
                float sv = s[(size_t)row * SDIM + sColOff + col];
#pragma unroll
                for (int t = 0; t < 10; ++t) {
                    float rr = rho_f(sv);
                    // 0.5*s + 0.5*rhod(s)*C, rhod = 4r(1-r)
                    sv = 0.5f * sv + 2.0f * rr * (1.0f - rr) * c;
                }
                out[(size_t)row * SDIM + outColOff + col] = sv;
                if (Anext != nullptr)
                    Anext[(size_t)row * NDIM + col] = (_Float16)rho_f(sv);
            }
        }
    }
}

extern "C" void kernel_launch(void* const* d_in, const int* in_sizes, int n_in,
                              void* d_out, int out_size, void* d_ws, size_t ws_size,
                              hipStream_t stream) {
    const float* Ux     = (const float*)d_in[0];
    const float* s      = (const float*)d_in[1];
    const float* W      = (const float*)d_in[2];
    const float* b_even = (const float*)d_in[3];
    const float* b_odd  = (const float*)d_in[4];
    float* out = (float*)d_out;

    char* ws = (char*)d_ws;
    const size_t szA = (size_t)B_ROWS * SE * sizeof(_Float16);  // 64 MB
    const size_t szW = (size_t)SE * SO * sizeof(_Float16);      //  8 MB
    _Float16* A1   = (_Float16*)(ws);
    _Float16* A2   = (_Float16*)(ws + szA);
    _Float16* W16  = (_Float16*)(ws + 2 * szA);
    _Float16* Wt16 = (_Float16*)(ws + 2 * szA + szW);

    // Phase 0 prep: A1 = f16(rho(s_even)); W -> f16 (and transposed f16)
    prep_rho_kernel<<<(B_ROWS * SE / 4) / 256, 256, 0, stream>>>(s, A1);
    prep_w_kernel<<<(SE * SO / 4) / 256, 256, 0, stream>>>(W, W16, Wt16);

    const int gemmGrid = (B_ROWS / TM) * (NDIM / TN);  // 128 * 16 = 2048

    // Phase 1: C_odd = A1 @ W + b_odd (+Ux), iterate s_odd, emit A2 = rho(s_odd)
    gemm_iter_kernel<<<gemmGrid, 256, 0, stream>>>(
        A1, Wt16, b_odd, Ux, s, out, A2, SE, SE);

    // Phase 2: C_even = A2 @ W^T + b_even, iterate s_even
    gemm_iter_kernel<<<gemmGrid, 256, 0, stream>>>(
        A2, W16, b_even, nullptr, s, out, nullptr, 0, 0);
}
